// HybridDynamicGraph_6305011990851
// MI455X (gfx1250) — matrix operand order repaired and checked by reading, not compiled, not run on hardware
//
#include <hip/hip_runtime.h>
#include <math.h>

#define N_ROWS 1024
#define HD     10000
#define HID    128
#define NTOT   256      // [U | V] columns
#define MTILE  128
#define KSTEP  32
#define KCHUNK 320      // multiple of KSTEP -> aligned, guard-free main path
#define KSPLIT ((HD + KCHUNK - 1) / KCHUNK)   // 32

typedef __attribute__((ext_vector_type(16))) __bf16 v16bf;
typedef __attribute__((ext_vector_type(8)))  __bf16 v8bf;
typedef __attribute__((ext_vector_type(4)))  __bf16 v4bf;
typedef __attribute__((ext_vector_type(8)))  float  v8f;
typedef __attribute__((ext_vector_type(4)))  float  v4f;

// ---------------------------------------------------------------------------
// Kernel 0: zero the UV accumulator workspace (split-K uses atomic adds).
// ---------------------------------------------------------------------------
__global__ void zero_uv(float* __restrict__ p, int n) {
    int i = blockIdx.x * blockDim.x + threadIdx.x;
    if (i < n) p[i] = 0.0f;
}

// ---------------------------------------------------------------------------
// Kernel 1: split-K bf16 WMMA GEMM with register double-buffered staging.
//   UV[m][n] += sum_k features[m][k] * Wt[k][n]
//   Wt[k][n] = w1[n & 127][(n >> 7) * HD + k]
// Block: 512 threads = 16 waves, wave grid 4(M) x 4(N); wave tile 32x64.
// ---------------------------------------------------------------------------
__global__ __launch_bounds__(512) void gemm_uv(const float* __restrict__ feat,
                                               const float* __restrict__ w1,
                                               float* __restrict__ UV)
{
    __shared__ __align__(16) __bf16 Alds[MTILE][40];   // 32 used cols + pad
    __shared__ __align__(16) __bf16 Blds[NTOT][40];

    const int tid   = threadIdx.x;
    const int mtile = blockIdx.x * MTILE;

    const int k_begin = blockIdx.y * KCHUNK;
    const int k_end   = (k_begin + KCHUNK < HD) ? (k_begin + KCHUNK) : HD;

    const int lane = tid & 31;
    const int wave = tid >> 5;
    const int wm   = wave >> 2;     // 0..3 -> M offset 32*wm
    const int wn   = wave & 3;      // 0..3 -> N offset 64*wn

    v8f acc[2][4];
    #pragma unroll
    for (int i = 0; i < 2; i++)
        #pragma unroll
        for (int j = 0; j < 4; j++)
            acc[i][j] = (v8f){0.f,0.f,0.f,0.f,0.f,0.f,0.f,0.f};

    const int lr = tid >> 3;         // 0..63 : row loaded per pass
    const int lc = (tid & 7) * 4;    // 0,4,...,28 : k offset (float4 granule)

    // staging registers: 2 A rows + 4 B rows, one float4 each
    v4f ra[2], rb[4];

    // row base pointers (invariant over k)
    const float* asrc[2];
    const float* bsrc[4];
    #pragma unroll
    for (int p = 0; p < 2; p++)
        asrc[p] = feat + (size_t)(mtile + lr + p * 64) * HD;
    #pragma unroll
    for (int p = 0; p < 4; p++) {
        int n = lr + p * 64;
        bsrc[p] = w1 + (size_t)(n & 127) * (2 * HD) + (size_t)(n >> 7) * HD;
    }

    auto load_tiles = [&](int k0) {
        if (k0 + KSTEP <= k_end) {
            // fast path: unguarded, 16B-aligned b128 loads
            #pragma unroll
            for (int p = 0; p < 2; p++)
                ra[p] = __builtin_nontemporal_load((const v4f*)(asrc[p] + k0 + lc));
            #pragma unroll
            for (int p = 0; p < 4; p++)
                rb[p] = *(const v4f*)(bsrc[p] + k0 + lc);
        } else {
            // tail path (once per block at most): per-element guard
            #pragma unroll
            for (int p = 0; p < 2; p++)
                #pragma unroll
                for (int i = 0; i < 4; i++) {
                    int kk = k0 + lc + i;
                    ra[p][i] = (kk < k_end) ? asrc[p][kk] : 0.0f;
                }
            #pragma unroll
            for (int p = 0; p < 4; p++)
                #pragma unroll
                for (int i = 0; i < 4; i++) {
                    int kk = k0 + lc + i;
                    rb[p][i] = (kk < k_end) ? bsrc[p][kk] : 0.0f;
                }
        }
    };

    const int nsteps = (k_end - k_begin + KSTEP - 1) / KSTEP;
    load_tiles(k_begin);                       // prefetch step 0

    for (int s = 0; s < nsteps; s++) {
        // ---- convert staged registers to bf16 and commit to LDS ----
        #pragma unroll
        for (int p = 0; p < 2; p++) {
            v4bf pk;
            #pragma unroll
            for (int i = 0; i < 4; i++) pk[i] = (__bf16)ra[p][i];
            *(v4bf*)&Alds[lr + p * 64][lc] = pk;
        }
        #pragma unroll
        for (int p = 0; p < 4; p++) {
            v4bf pk;
            #pragma unroll
            for (int i = 0; i < 4; i++) pk[i] = (__bf16)rb[p][i];
            *(v4bf*)&Blds[lr + p * 64][lc] = pk;
        }
        __syncthreads();

        // ---- prefetch next step's globals while WMMAs run ----
        if (s + 1 < nsteps) load_tiles(k_begin + (s + 1) * KSTEP);

        // ---- fragments per ISA 16-bit A/B layouts ----
        const int arow0 = wm * 32 + (lane & 15);
        const int akb   = (lane >> 4) * 8;          // lanes 0-15: K 0..7 / 16..23
        v16bf afrag[2];
        #pragma unroll
        for (int ms = 0; ms < 2; ms++) {
            v8bf lo = *(const v8bf*)&Alds[arow0 + ms * 16][akb];
            v8bf hi = *(const v8bf*)&Alds[arow0 + ms * 16][16 + akb];
            afrag[ms] = __builtin_shufflevector(lo, hi,
                0,1,2,3,4,5,6,7,8,9,10,11,12,13,14,15);
        }
        const int brow0 = wn * 64 + (lane & 15);
        const int bkb   = (lane >> 4) * 8;          // lanes 0-15: K 0..7 / 16..23
        #pragma unroll
        for (int ns = 0; ns < 4; ns++) {
            v8bf lo = *(const v8bf*)&Blds[brow0 + ns * 16][bkb];
            v8bf hi = *(const v8bf*)&Blds[brow0 + ns * 16][16 + bkb];
            v16bf bfrag = __builtin_shufflevector(lo, hi,
                0,1,2,3,4,5,6,7,8,9,10,11,12,13,14,15);
            #pragma unroll
            for (int ms = 0; ms < 2; ms++)
                acc[ms][ns] = __builtin_amdgcn_wmma_f32_16x16x32_bf16(
                    false, afrag[ms], false, bfrag,
                    (short)0, acc[ms][ns], false, false);
        }
        __syncthreads();
    }

    // ---- split-K partial accumulate: C layout lane l, vgpr v ->
    //      (M = 8*(l>>4)+v, N = l&15) within each 16x16 tile ----
    #pragma unroll
    for (int ms = 0; ms < 2; ms++)
        #pragma unroll
        for (int ns = 0; ns < 4; ns++)
            #pragma unroll
            for (int v = 0; v < 8; v++) {
                int row = mtile + wm * 32 + ms * 16 + (lane >> 4) * 8 + v;
                int col = wn * 64 + ns * 16 + (lane & 15);
                atomicAdd(&UV[(size_t)row * NTOT + col], acc[ms][ns][v]);
            }
}

// ---------------------------------------------------------------------------
// Kernel 2: pairwise relu/sigmoid combine over upper-triangular 32x32 blocks.
// Row stride 132 floats = 528B (16B aligned) -> b128 LDS reads; Vt/b1/w2 reads
// shared across the 4 pairs each lane owns.
// ---------------------------------------------------------------------------
__global__ __launch_bounds__(256) void pairwise(const float* __restrict__ UV,
                                                const float* __restrict__ b1,
                                                const float* __restrict__ w2,
                                                const float* __restrict__ b2,
                                                float* __restrict__ out)
{
    const int bj = blockIdx.x;
    const int bi = blockIdx.y;
    if (bi > bj) return;                       // only upper-triangular blocks

    __shared__ __align__(16) float Ut[32][132];
    __shared__ __align__(16) float Vt[32][132];
    __shared__ __align__(16) float b1s[HID];
    __shared__ __align__(16) float w2s[HID];

    const int tid = threadIdx.x;
    for (int idx = tid; idx < 32 * HID; idx += 256) {
        int r = idx >> 7, c = idx & 127;
        Ut[r][c] = UV[(size_t)(bi * 32 + r) * NTOT + c];
        Vt[r][c] = UV[(size_t)(bj * 32 + r) * NTOT + HID + c];
    }
    if (tid < HID) { b1s[tid] = b1[tid]; w2s[tid] = w2[tid]; }
    __syncthreads();

    const float bias2 = b2[0];
    const int jj = tid & 31;
    const int i0 = tid >> 5;

    float s[4];
    #pragma unroll
    for (int p = 0; p < 4; p++) s[p] = bias2;

    for (int h = 0; h < HID; h += 4) {
        v4f vv = *(const v4f*)&Vt[jj][h];
        v4f bb = *(const v4f*)&b1s[h];
        v4f ww = *(const v4f*)&w2s[h];
        #pragma unroll
        for (int p = 0; p < 4; p++) {
            v4f uu = *(const v4f*)&Ut[i0 + p * 8][h];
            #pragma unroll
            for (int e = 0; e < 4; e++)
                s[p] = fmaf(ww[e], fmaxf(uu[e] + vv[e] + bb[e], 0.0f), s[p]);
        }
    }

    #pragma unroll
    for (int p = 0; p < 4; p++) {
        int ii = i0 + p * 8;
        int gi = bi * 32 + ii;
        int gj = bj * 32 + jj;
        if (bi == bj && ii > jj) continue;     // mirror covers lower half
        float sig = 1.0f / (1.0f + __expf(-s[p]));
        if (gi == gj) {
            out[(size_t)gi * N_ROWS + gj] = 0.0f;
        } else {
            out[(size_t)gi * N_ROWS + gj] = sig;
            out[(size_t)gj * N_ROWS + gi] = sig;
        }
    }
}

// ---------------------------------------------------------------------------
extern "C" void kernel_launch(void* const* d_in, const int* in_sizes, int n_in,
                              void* d_out, int out_size, void* d_ws, size_t ws_size,
                              hipStream_t stream) {
    const float* feat = (const float*)d_in[0];   // [1024, 10000]
    const float* w1   = (const float*)d_in[1];   // [128, 20000]
    const float* b1   = (const float*)d_in[2];   // [128]
    const float* w2   = (const float*)d_in[3];   // [1, 128]
    const float* b2   = (const float*)d_in[4];   // [1]
    float* out = (float*)d_out;                  // [1024, 1024]
    float* UV  = (float*)d_ws;                   // [1024, 256] f32 partials

    const int uv_elems = N_ROWS * NTOT;
    zero_uv<<<(uv_elems + 255) / 256, 256, 0, stream>>>(UV, uv_elems);
    gemm_uv<<<dim3(N_ROWS / MTILE, KSPLIT), 512, 0, stream>>>(feat, w1, UV);
    pairwise<<<dim3(32, 32), 256, 0, stream>>>(UV, b1, w2, b2, out);
}
